// AdversarialAttack_85993835200845
// MI455X (gfx1250) — matrix operand order repaired and checked by reading, not compiled, not run on hardware
//
#include <hip/hip_runtime.h>

// ---------------------------------------------------------------------------
// Problem constants (reference: B=4, S=2048, V=32000, D=2048, N_ATK=20)
// ---------------------------------------------------------------------------
#define BATCH   4
#define SEQ     2048
#define VOCAB   32000
#define DIM     2048
#define MROWS   (BATCH * SEQ)          // 8192 GEMM rows

// GEMM tiling
#define M_TILE  128
#define N_TILE  128
#define KC      32                      // K chunk staged in LDS per step
#define LSTR    40                      // padded LDS row stride (bf16 elems) -> conflict-free b128 reads
#define NSPLIT  10                      // N-range splits (grid.y)
#define NT_PER  ((VOCAB / N_TILE) / NSPLIT)   // 25 n-tiles per split

typedef __attribute__((ext_vector_type(8)))  __bf16 bf16x8;
typedef __attribute__((ext_vector_type(16))) __bf16 bf16x16;
typedef __attribute__((ext_vector_type(8)))  float  v8f;

// ---------------------------------------------------------------------------
// fp32 -> bf16 hi/lo split (round-to-nearest-even)
// ---------------------------------------------------------------------------
__device__ __forceinline__ unsigned short bf16_rne(float f) {
    unsigned int u = __builtin_bit_cast(unsigned int, f);
    u += 0x7FFFu + ((u >> 16) & 1u);
    return (unsigned short)(u >> 16);
}
__device__ __forceinline__ float bf16_to_f32(unsigned short h) {
    unsigned int u = ((unsigned int)h) << 16;
    return __builtin_bit_cast(float, u);
}
__device__ __forceinline__ void split4(float4 v, ushort4& h, ushort4& l) {
    h.x = bf16_rne(v.x); l.x = bf16_rne(v.x - bf16_to_f32(h.x));
    h.y = bf16_rne(v.y); l.y = bf16_rne(v.y - bf16_to_f32(h.y));
    h.z = bf16_rne(v.z); l.z = bf16_rne(v.z - bf16_to_f32(h.z));
    h.w = bf16_rne(v.w); l.w = bf16_rne(v.w - bf16_to_f32(h.w));
}

// two 16-byte LDS reads -> one 16-element bf16 WMMA fragment
__device__ __forceinline__ bf16x16 ldfrag(const unsigned short* p0,
                                          const unsigned short* p1) {
    bf16x8 a = *(const bf16x8*)p0;
    bf16x8 b = *(const bf16x8*)p1;
    return __builtin_shufflevector(a, b, 0,1,2,3,4,5,6,7,8,9,10,11,12,13,14,15);
}

__device__ __forceinline__ v8f wmma_bf16(bf16x16 a, bf16x16 b, v8f c) {
    return __builtin_amdgcn_wmma_f32_16x16x32_bf16(
        /*neg_a=*/false, a, /*neg_b=*/false, b,
        /*c_mod=*/(short)0, c, /*reuse_a=*/false, /*reuse_b=*/false);
}

// CDNA5 async copy: global -> LDS, 16 bytes per lane, tracked by ASYNCcnt.
// lds_off = low 32 bits of the generic LDS address (aperture rule: addr[31:0]).
__device__ __forceinline__ void async_ld_b128(unsigned lds_off, const void* g) {
    asm volatile("global_load_async_to_lds_b128 %0, %1, off"
                 :: "v"(lds_off), "v"((unsigned long long)(uintptr_t)g)
                 : "memory");
}
__device__ __forceinline__ void wait_async0() {
    asm volatile("s_wait_asynccnt 0x0" ::: "memory");
}

// ---------------------------------------------------------------------------
// Kernel 1: build inputs_embeds (gather + masked scatter of tiled attack rows)
// AND produce the bf16 hi/lo split of each row (single pass). one block / row.
// ---------------------------------------------------------------------------
__global__ __launch_bounds__(256) void embed_build_split_kernel(
    const int* __restrict__ input_ids, const unsigned char* __restrict__ mask,
    const float* __restrict__ attack, const float* __restrict__ emb,
    float* __restrict__ out, unsigned short* __restrict__ Ah,
    unsigned short* __restrict__ Al, int n_atk)
{
    const int bs = blockIdx.x;
    const int b  = bs >> 11;           // /SEQ
    const int s  = bs & (SEQ - 1);
    const unsigned char* mrow = mask + (size_t)b * SEQ;

    // rank = (#masked positions in [0..s]) - 1  (general mask scan)
    int c = 0;
    for (int x = threadIdx.x; x <= s; x += 256) c += mrow[x] ? 1 : 0;
    #pragma unroll
    for (int m = 16; m >= 1; m >>= 1) c += __shfl_xor(c, m, 32);
    __shared__ int cs;
    if (threadIdx.x == 0) cs = 0;
    __syncthreads();
    if ((threadIdx.x & 31) == 0) atomicAdd(&cs, c);
    __syncthreads();

    const float* src;
    if (mrow[s]) {
        const int rank = (cs - 1) % n_atk;
        src = attack + (size_t)rank * DIM;
    } else {
        src = emb + (size_t)input_ids[bs] * DIM;
    }
    const float4* s4 = (const float4*)src;
    float4*       d4 = (float4*)(out + (size_t)bs * DIM);
    unsigned short* ah = Ah + (size_t)bs * DIM;
    unsigned short* al = Al + (size_t)bs * DIM;
    for (int k = threadIdx.x; k < DIM / 4; k += 256) {
        const float4 v = s4[k];
        d4[k] = v;
        ushort4 h, l;
        split4(v, h, l);
        *(ushort4*)(ah + k * 4) = h;
        *(ushort4*)(al + k * 4) = l;
    }
}

// ---------------------------------------------------------------------------
// Kernel 2: single pass over emb_weight: bf16 hi/lo split + w2[v] = ||w_v||^2
// ---------------------------------------------------------------------------
__global__ __launch_bounds__(256) void wsplit_w2_kernel(
    const float* __restrict__ W, unsigned short* __restrict__ Wh,
    unsigned short* __restrict__ Wl, float* __restrict__ w2)
{
    const int v = blockIdx.x;
    const float4* s4 = (const float4*)(W + (size_t)v * DIM);
    unsigned short* wh = Wh + (size_t)v * DIM;
    unsigned short* wl = Wl + (size_t)v * DIM;
    float ss = 0.f;
    for (int k = threadIdx.x; k < DIM / 4; k += 256) {
        const float4 x = s4[k];
        ss += x.x * x.x + x.y * x.y + x.z * x.z + x.w * x.w;
        ushort4 h, l;
        split4(x, h, l);
        *(ushort4*)(wh + k * 4) = h;
        *(ushort4*)(wl + k * 4) = l;
    }
    #pragma unroll
    for (int m = 16; m >= 1; m >>= 1) ss += __shfl_xor(ss, m, 32);
    __shared__ float red[8];
    if ((threadIdx.x & 31) == 0) red[threadIdx.x >> 5] = ss;
    __syncthreads();
    if (threadIdx.x == 0) {
        float t = 0.f;
        #pragma unroll
        for (int i = 0; i < 8; ++i) t += red[i];
        w2[v] = t;
    }
}

// ---------------------------------------------------------------------------
// Kernel 3: WMMA GEMM (split-bf16 x3 for ~fp32 accuracy) + running argmin.
// grid = (MROWS/M_TILE, NSPLIT), block = 256 (8 waves).
// Staging is async copy global->LDS (no VGPR round trip, ASYNCcnt-tracked).
// Wave w owns rows [w*16, w*16+16) of the block's M tile and the full N tile.
// ---------------------------------------------------------------------------
__global__ __launch_bounds__(256) void gemm_argmin_kernel(
    const unsigned short* __restrict__ Ah, const unsigned short* __restrict__ Al,
    const unsigned short* __restrict__ Wh, const unsigned short* __restrict__ Wl,
    const float* __restrict__ w2,
    float* __restrict__ pval,         // [NSPLIT][MROWS] partial min d2
    int*   __restrict__ pidx)         // [NSPLIT][MROWS] partial argmin
{
    __shared__ __align__(16) unsigned short sAh[M_TILE * LSTR];
    __shared__ __align__(16) unsigned short sAl[M_TILE * LSTR];
    __shared__ __align__(16) unsigned short sBh[N_TILE * LSTR];
    __shared__ __align__(16) unsigned short sBl[N_TILE * LSTR];

    const int t    = threadIdx.x;
    const int lane = t & 31;
    const int wid  = t >> 5;
    const int lr   = lane & 15;        // N-col / M-row within fragment
    const int hi   = lane >> 4;        // half-wave select
    const int m0   = blockIdx.x * M_TILE;

    // LDS byte offsets (generic address low 32 bits == LDS offset)
    const unsigned oAh = (unsigned)(uintptr_t)&sAh[0];
    const unsigned oAl = (unsigned)(uintptr_t)&sAl[0];
    const unsigned oBh = (unsigned)(uintptr_t)&sBh[0];
    const unsigned oBl = (unsigned)(uintptr_t)&sBl[0];

    // A fragment source rows for this wave (lane holds row lr;
    // low half K = [0,8)+[16,24), high half K = [8,16)+[24,32))
    const unsigned short* aRH = sAh + (wid * 16 + lr) * LSTR;
    const unsigned short* aRL = sAl + (wid * 16 + lr) * LSTR;
    const int ca = hi ? 8 : 0;         // A per-half K offset
    const int cb = hi ? 8 : 0;         // B per-half K offset

    float minv[8];
    int   mini[8];
    #pragma unroll
    for (int r = 0; r < 8; ++r) { minv[r] = 3.4e38f; mini[r] = 0; }

    for (int it = 0; it < NT_PER; ++it) {
        const int nt = blockIdx.y * NT_PER + it;
        const int n0 = nt * N_TILE;

        v8f acc[8];
        #pragma unroll
        for (int j = 0; j < 8; ++j)
            acc[j] = (v8f){0.f, 0.f, 0.f, 0.f, 0.f, 0.f, 0.f, 0.f};

        for (int kc = 0; kc < DIM / KC; ++kc) {
            const int kb = kc * KC;
            __syncthreads();   // previous chunk fully consumed before overwrite
            // 512 x 16B chunks per array; 2 per thread per array
            #pragma unroll
            for (int rep = 0; rep < 2; ++rep) {
                const int c   = rep * 256 + t;
                const int row = c >> 2;          // 0..127
                const int seg = c & 3;           // 16B segment in the 64B row chunk
                const size_t goffA = (size_t)(m0 + row) * DIM + kb + seg * 8;
                const size_t goffB = (size_t)(n0 + row) * DIM + kb + seg * 8;
                const unsigned loff = (unsigned)(row * (LSTR * 2) + seg * 16);
                async_ld_b128(oAh + loff, Ah + goffA);
                async_ld_b128(oAl + loff, Al + goffA);
                async_ld_b128(oBh + loff, Wh + goffB);
                async_ld_b128(oBl + loff, Wl + goffB);
                if (kc + 1 < DIM / KC)
                    __builtin_prefetch((const void*)(Wh + goffB + KC), 0, 3);
            }
            wait_async0();     // own async writes to LDS landed
            __syncthreads();   // everyone's landed

            const bf16x16 Ahf = ldfrag(aRH + ca, aRH + ca + 16);
            const bf16x16 Alf = ldfrag(aRL + ca, aRL + ca + 16);
            #pragma unroll
            for (int j = 0; j < 8; ++j) {
                const unsigned short* bRH = sBh + (j * 16 + lr) * LSTR;
                const unsigned short* bRL = sBl + (j * 16 + lr) * LSTR;
                const bf16x16 Bhf = ldfrag(bRH + cb, bRH + cb + 16);
                const bf16x16 Blf = ldfrag(bRL + cb, bRL + cb + 16);
                acc[j] = wmma_bf16(Ahf, Bhf, acc[j]);  // hi*hi
                acc[j] = wmma_bf16(Ahf, Blf, acc[j]);  // hi*lo
                acc[j] = wmma_bf16(Alf, Bhf, acc[j]);  // lo*hi
            }
        }

        // d2 = w2[n] - 2*cross  (x2 constant per row: irrelevant for argmin)
        #pragma unroll
        for (int j = 0; j < 8; ++j) {
            const int n = n0 + j * 16 + lr;
            const float w2n = w2[n];
            #pragma unroll
            for (int r = 0; r < 8; ++r) {
                const float d2 = w2n - 2.0f * acc[j][r];
                if (d2 < minv[r]) { minv[r] = d2; mini[r] = n; }  // first-index ties
            }
        }
    }

    // cross-lane argmin within each 16-lane half (lanes share a row, differ in n)
    #pragma unroll
    for (int r = 0; r < 8; ++r) {
        float v = minv[r];
        int   i = mini[r];
        #pragma unroll
        for (int m = 8; m >= 1; m >>= 1) {
            const float ov = __shfl_xor(v, m, 32);
            const int   oi = __shfl_xor(i, m, 32);
            if (ov < v || (ov == v && oi < i)) { v = ov; i = oi; }
        }
        minv[r] = v; mini[r] = i;
    }
    if (lr == 0) {
        const int    mrow = m0 + wid * 16 + hi * 8;  // C-layout: VGPR r -> row r (+8 hi half)
        const size_t pb   = (size_t)blockIdx.y * MROWS;
        #pragma unroll
        for (int r = 0; r < 8; ++r) {
            pval[pb + mrow + r] = minv[r];
            pidx[pb + mrow + r] = mini[r];
        }
    }
}

// ---------------------------------------------------------------------------
// Kernel 4: merge NSPLIT partial argmins, emit ids as float
// ---------------------------------------------------------------------------
__global__ __launch_bounds__(256) void argmin_final_kernel(
    const float* __restrict__ pv, const int* __restrict__ pi,
    float* __restrict__ out_ids)
{
    const int m = blockIdx.x * 256 + threadIdx.x;
    float bv = pv[m];
    int   bi = pi[m];
    #pragma unroll
    for (int p = 1; p < NSPLIT; ++p) {
        const float v = pv[(size_t)p * MROWS + m];
        const int   i = pi[(size_t)p * MROWS + m];
        if (v < bv || (v == bv && i < bi)) { bv = v; bi = i; }
    }
    out_ids[m] = (float)bi;
}

// ---------------------------------------------------------------------------
// launcher
// workspace layout (~330 MB):
//   w2[VOCAB] | pval[NSPLIT*MROWS] | pidx[NSPLIT*MROWS] |
//   Ah[MROWS*DIM] | Al[MROWS*DIM] | Wh[VOCAB*DIM] | Wl[VOCAB*DIM]   (bf16)
// ---------------------------------------------------------------------------
extern "C" void kernel_launch(void* const* d_in, const int* in_sizes, int n_in,
                              void* d_out, int out_size, void* d_ws, size_t ws_size,
                              hipStream_t stream) {
    const int*           input_ids = (const int*)d_in[0];
    const unsigned char* mask      = (const unsigned char*)d_in[1];
    const float*         attack    = (const float*)d_in[2];
    const float*         emb       = (const float*)d_in[3];

    const int n_atk = in_sizes[2] / DIM;

    float* out_emb = (float*)d_out;                          // [MROWS][DIM]
    float* out_ids = (float*)d_out + (size_t)MROWS * DIM;    // [MROWS]

    float*          w2 = (float*)d_ws;
    float*          pv = w2 + VOCAB;
    int*            pi = (int*)(pv + (size_t)NSPLIT * MROWS);
    unsigned short* Ah = (unsigned short*)(pi + (size_t)NSPLIT * MROWS);
    unsigned short* Al = Ah + (size_t)MROWS * DIM;
    unsigned short* Wh = Al + (size_t)MROWS * DIM;
    unsigned short* Wl = Wh + (size_t)VOCAB * DIM;

    embed_build_split_kernel<<<MROWS, 256, 0, stream>>>(input_ids, mask, attack,
                                                        emb, out_emb, Ah, Al, n_atk);
    wsplit_w2_kernel<<<VOCAB, 256, 0, stream>>>(emb, Wh, Wl, w2);
    gemm_argmin_kernel<<<dim3(MROWS / M_TILE, NSPLIT), 256, 0, stream>>>(
        Ah, Al, Wh, Wl, w2, pv, pi);
    argmin_final_kernel<<<MROWS / 256, 256, 0, stream>>>(pv, pi, out_ids);
}
